// GRN_GAT_Bidirectional_5506148073982
// MI455X (gfx1250) — compile-verified, matrix operand order checked by reading, not run on hardware
//
#include <hip/hip_runtime.h>
#include <hip/hip_bf16.h>

// ---------------------------------------------------------------------------
// GRN GAT (2-layer, bidirectional + self-loops) + pair-MLP link predictor.
// Dense GEMMs run on CDNA5 WMMA (v_wmma_f32_16x16x32_f16, wave32).
// Irregular edge work (softmax over segments, scatter-add) uses atomics.
// ---------------------------------------------------------------------------

typedef _Float16 v8h  __attribute__((ext_vector_type(8)));
typedef _Float16 v16h __attribute__((ext_vector_type(16)));
typedef float    v8f  __attribute__((ext_vector_type(8)));

#define N_NODES 20000
#define E_EDGES 200000
#define FN_DIM  128
#define FE_DIM  8
#define H1_HEADS 4
#define C_CH    64
#define HC1     (H1_HEADS * C_CH)   // 256
#define E2_EDGES (2 * E_EDGES)      // 400000
#define ETOT    (E2_EDGES + N_NODES) // 420000

// ---- order-preserving float<->uint encoding for atomicMax-based segment max
static __device__ __forceinline__ unsigned ordEncode(float f) {
  unsigned u = __float_as_uint(f);
  return (u & 0x80000000u) ? ~u : (u | 0x80000000u);
}
static __device__ __forceinline__ float ordDecode(unsigned u) {
  return __uint_as_float((u & 0x80000000u) ? (u & 0x7FFFFFFFu) : ~u);
}

// ---------------------------------------------------------------------------
// Graph construction: bidirectional edges + per-node mean edge_attr (for loops)
// ---------------------------------------------------------------------------
__global__ void build_edges_kernel(const int* __restrict__ ei,
                                   const float* __restrict__ edge_attr,
                                   int* __restrict__ src, int* __restrict__ dst,
                                   float* __restrict__ ea,
                                   float* __restrict__ cnt,
                                   float* __restrict__ easum) {
  int e = blockIdx.x * blockDim.x + threadIdx.x;
  if (e >= E2_EDGES) return;
  // src2 = concat(ei[0], ei[1]) == ei_flat[e]; dst2 = concat(ei[1], ei[0])
  int s = ei[e];
  int d = (e < E_EDGES) ? ei[E_EDGES + e] : ei[e - E_EDGES];
  src[e] = s;
  dst[e] = d;
  atomicAdd(&cnt[d], 1.0f);
  int esrc = (e < E_EDGES) ? e : (e - E_EDGES);
#pragma unroll
  for (int f = 0; f < FE_DIM; ++f) {
    float v = edge_attr[esrc * FE_DIM + f];
    ea[(size_t)e * FE_DIM + f] = v;
    atomicAdd(&easum[d * FE_DIM + f], v);
  }
}

__global__ void selfloop_kernel(int* __restrict__ src, int* __restrict__ dst,
                                float* __restrict__ ea,
                                const float* __restrict__ cnt,
                                const float* __restrict__ easum) {
  int n = blockIdx.x * blockDim.x + threadIdx.x;
  if (n >= N_NODES) return;
  src[E2_EDGES + n] = n;
  dst[E2_EDGES + n] = n;
  float c = fmaxf(cnt[n], 1.0f);
#pragma unroll
  for (int f = 0; f < FE_DIM; ++f)
    ea[(size_t)(E2_EDGES + n) * FE_DIM + f] = easum[n * FE_DIM + f] / c;
}

// ---------------------------------------------------------------------------
// Conversions
// ---------------------------------------------------------------------------
__global__ void cvt_f32_to_f16_kernel(const float* __restrict__ in,
                                      _Float16* __restrict__ out, long total) {
  long i = (long)blockIdx.x * blockDim.x + threadIdx.x;
  if (i < total) out[i] = (_Float16)in[i];
}

// W[K][Nc] (row-major f32)  ->  Wt[Nc][K] (row-major f16)
__global__ void transpose_cvt_kernel(const float* __restrict__ W,
                                     _Float16* __restrict__ Wt, int K, int Nc) {
  int i = blockIdx.x * blockDim.x + threadIdx.x;
  if (i >= K * Nc) return;
  int k = i / Nc, n = i % Nc;
  Wt[(size_t)n * K + k] = (_Float16)W[i];
}

// ---------------------------------------------------------------------------
// WMMA GEMM: C[M][Nc] (f32) = A[M][K] (f16, row-major) @ Bt[Nc][K]^T (f16)
// One wave per 16x16 output tile. K multiple of 32, M/Nc multiples of 16.
// ---------------------------------------------------------------------------
__global__ void gemm_wmma_kernel(const _Float16* __restrict__ A,
                                 const _Float16* __restrict__ Bt,
                                 float* __restrict__ C,
                                 int M, int K, int Nc) {
  int wave = blockIdx.x * (blockDim.x >> 5) + (threadIdx.x >> 5);
  int lane = threadIdx.x & 31;
  int ntiles = Nc >> 4;
  int totalTiles = (M >> 4) * ntiles;
  if (wave >= totalTiles) return;               // wave-uniform: EXEC stays full
  int mtile = wave / ntiles, ntile = wave % ntiles;
  int idx16 = lane & 15, half = lane >> 4;

  const _Float16* arow = A + (size_t)(mtile * 16 + idx16) * K;
  const _Float16* brow = Bt + (size_t)(ntile * 16 + idx16) * K;

  v8f acc = {};
  for (int kk = 0; kk < K; kk += 32) {
    int k0 = kk + 8 * half;        // elements 0..7 of the fragment
    int k1 = kk + 16 + 8 * half;   // elements 8..15
    v8h a_lo = *(const v8h*)(arow + k0);
    v8h a_hi = *(const v8h*)(arow + k1);
    v8h b_lo = *(const v8h*)(brow + k0);
    v8h b_hi = *(const v8h*)(brow + k1);
    v16h a = __builtin_shufflevector(a_lo, a_hi, 0,1,2,3,4,5,6,7,8,9,10,11,12,13,14,15);
    v16h b = __builtin_shufflevector(b_lo, b_hi, 0,1,2,3,4,5,6,7,8,9,10,11,12,13,14,15);
    acc = __builtin_amdgcn_wmma_f32_16x16x32_f16(false, a, false, b,
                                                 (short)0, acc, false, false);
  }
  // C/D layout: n = lane%16, m = r + 8*(lane/16)
  float* crow = C + (size_t)(mtile * 16 + 8 * half) * Nc + ntile * 16 + idx16;
#pragma unroll
  for (int r = 0; r < 8; ++r) crow[(size_t)r * Nc] = acc[r];
}

// ---------------------------------------------------------------------------
// Attention pieces
// ---------------------------------------------------------------------------
// al[n,h] = <xh[n,h,:], att_src[h,:]> ; ar likewise with att_dst
__global__ void attn_node_kernel(const float* __restrict__ xh,
                                 const float* __restrict__ att_src,
                                 const float* __restrict__ att_dst,
                                 float* __restrict__ al, float* __restrict__ ar,
                                 int H) {
  int i = blockIdx.x * blockDim.x + threadIdx.x;
  if (i >= N_NODES * H) return;
  int n = i / H, h = i % H;
  const float* xr = xh + (size_t)n * H * C_CH + h * C_CH;
  float sa = 0.f, sd = 0.f;
#pragma unroll 8
  for (int c = 0; c < C_CH; ++c) {
    float v = xr[c];
    sa += v * att_src[h * C_CH + c];
    sd += v * att_dst[h * C_CH + c];
  }
  al[i] = sa;
  ar[i] = sd;
}

// M[f,h] = sum_c We[f, h*C+c] * att_edge[h,c]   (FE x H, tiny)
__global__ void edge_att_matrix_kernel(const float* __restrict__ We,
                                       const float* __restrict__ att_edge,
                                       float* __restrict__ Mout, int H) {
  int i = threadIdx.x;
  if (i >= FE_DIM * H) return;
  int f = i / H, h = i % H;
  float s = 0.f;
  for (int c = 0; c < C_CH; ++c)
    s += We[(size_t)f * H * C_CH + h * C_CH + c] * att_edge[h * C_CH + c];
  Mout[f * H + h] = s;
}

__global__ void init_amax_kernel(unsigned* __restrict__ amax, int total) {
  int i = blockIdx.x * blockDim.x + threadIdx.x;
  if (i < total) amax[i] = ordEncode(-__builtin_inff());
}

// alpha = leaky_relu(al[src]+ar[dst]+ea@M, 0.2); track segment max per dst
__global__ void alpha_kernel(const int* __restrict__ src, const int* __restrict__ dst,
                             const float* __restrict__ ea,
                             const float* __restrict__ al, const float* __restrict__ ar,
                             const float* __restrict__ M,
                             float* __restrict__ alpha, unsigned* __restrict__ amax,
                             int H) {
  long i = (long)blockIdx.x * blockDim.x + threadIdx.x;
  if (i >= (long)ETOT * H) return;
  int e = (int)(i / H), h = (int)(i % H);
  int s = src[e], d = dst[e];
  float ae = 0.f;
#pragma unroll
  for (int f = 0; f < FE_DIM; ++f) ae += ea[(size_t)e * FE_DIM + f] * M[f * H + h];
  float a = al[s * H + h] + ar[d * H + h] + ae;
  a = (a > 0.f) ? a : 0.2f * a;                 // leaky_relu(0.2)
  alpha[i] = a;
  atomicMax(&amax[d * H + h], ordEncode(a));
}

// p = exp(alpha - amax[dst]); den[dst] += p  (p overwrites alpha)
__global__ void exp_den_kernel(const int* __restrict__ dst,
                               float* __restrict__ alpha,
                               const unsigned* __restrict__ amax,
                               float* __restrict__ den, int H) {
  long i = (long)blockIdx.x * blockDim.x + threadIdx.x;
  if (i >= (long)ETOT * H) return;
  int e = (int)(i / H), h = (int)(i % H);
  float p = __expf(alpha[i] - ordDecode(amax[dst[e] * H + h]));
  alpha[i] = p;
  atomicAdd(&den[dst[e] * H + h], p);
}

// out[dst, h, c] += xh[src, h, c] * p / (den[dst,h] + 1e-16)
__global__ void aggregate_kernel(const int* __restrict__ src, const int* __restrict__ dst,
                                 const float* __restrict__ xh,
                                 const float* __restrict__ p,
                                 const float* __restrict__ den,
                                 float* __restrict__ out, int H) {
  int HC = H * C_CH;
  long i = (long)blockIdx.x * blockDim.x + threadIdx.x;
  if (i >= (long)ETOT * HC) return;
  int e = (int)(i / HC), j = (int)(i % HC);
  int h = j / C_CH;
  int s = src[e], d = dst[e];
  float w = p[(size_t)e * H + h] / (den[d * H + h] + 1e-16f);
  atomicAdd(&out[(size_t)d * HC + j], xh[(size_t)s * HC + j] * w);
}

// h = elu(out + b); also emit f16 copy for the next WMMA GEMM
__global__ void bias_elu_cvt_kernel(const float* __restrict__ in,
                                    const float* __restrict__ b,
                                    _Float16* __restrict__ out_h,
                                    int HC, long total) {
  long i = (long)blockIdx.x * blockDim.x + threadIdx.x;
  if (i >= total) return;
  float v = in[i] + b[i % HC];
  v = (v > 0.f) ? v : expm1f(v);                // elu(alpha=1)
  out_h[i] = (_Float16)v;
}

// ---------------------------------------------------------------------------
// Fused classifier: per 16-pair tile (one wave):
//   emb = [h2[p0], h2[p1]] (16x128, gathered) ; hid = relu(emb @ Wc1 + bc1)
//   logit = hid @ Wc2 + bc2  (cross-lane reduction over the 64 hidden units)
// ---------------------------------------------------------------------------
__global__ void classifier_kernel(const int* __restrict__ pairs,
                                  const _Float16* __restrict__ h2,
                                  const _Float16* __restrict__ Wc1t, // [64][128]
                                  const float* __restrict__ bc1,
                                  const float* __restrict__ Wc2,
                                  const float* __restrict__ bc2,
                                  float* __restrict__ out) {
  int wave = blockIdx.x * (blockDim.x >> 5) + (threadIdx.x >> 5);
  int lane = threadIdx.x & 31;
  if (wave >= E_EDGES / 16) return;             // wave-uniform
  int idx16 = lane & 15, half = lane >> 4;
  int m = wave * 16 + idx16;                    // A-fragment row for this lane
  int p0 = pairs[2 * m], p1 = pairs[2 * m + 1];
  const _Float16* r0 = h2 + (size_t)p0 * C_CH;  // first 64 of emb row
  const _Float16* r1 = h2 + (size_t)p1 * C_CH;  // last 64 of emb row

  float part[8];
#pragma unroll
  for (int r = 0; r < 8; ++r) part[r] = 0.f;

#pragma unroll
  for (int ntile = 0; ntile < 4; ++ntile) {
    const _Float16* brow = Wc1t + (size_t)(ntile * 16 + idx16) * 128;
    v8f acc = {};
#pragma unroll
    for (int s = 0; s < 4; ++s) {
      int k0 = 32 * s + 8 * half;
      int k1 = k0 + 16;
      // k-range per step is uniform w.r.t. the p0/p1 boundary (s<2 -> p0)
      const _Float16* srcA = (s < 2) ? r0 : r1;
      int kadj = (s < 2) ? 0 : 64;
      v8h a_lo = *(const v8h*)(srcA + k0 - kadj);
      v8h a_hi = *(const v8h*)(srcA + k1 - kadj);
      v8h b_lo = *(const v8h*)(brow + k0);
      v8h b_hi = *(const v8h*)(brow + k1);
      v16h a = __builtin_shufflevector(a_lo, a_hi, 0,1,2,3,4,5,6,7,8,9,10,11,12,13,14,15);
      v16h b = __builtin_shufflevector(b_lo, b_hi, 0,1,2,3,4,5,6,7,8,9,10,11,12,13,14,15);
      acc = __builtin_amdgcn_wmma_f32_16x16x32_f16(false, a, false, b,
                                                   (short)0, acc, false, false);
    }
    int n = ntile * 16 + idx16;                 // hidden unit index of this lane
    float bias = bc1[n], w2 = Wc2[n];
#pragma unroll
    for (int r = 0; r < 8; ++r) {
      float v = acc[r] + bias;                  // acc[r] is hid[m=r+8*half][n]
      part[r] += ((v > 0.f) ? v : 0.f) * w2;
    }
  }
  // reduce over the 16 lanes (= hidden-unit columns) within each half-wave
#pragma unroll
  for (int off = 1; off < 16; off <<= 1)
#pragma unroll
    for (int r = 0; r < 8; ++r) part[r] += __shfl_xor(part[r], off, 32);

  if (idx16 == 0) {
    float b2 = bc2[0];
    int mbase = wave * 16 + 8 * half;
#pragma unroll
    for (int r = 0; r < 8; ++r) out[mbase + r] = part[r] + b2;
  }
}

// ---------------------------------------------------------------------------
// Host launcher
// ---------------------------------------------------------------------------
static inline unsigned gridFor(long total, int block) {
  return (unsigned)((total + block - 1) / block);
}

extern "C" void kernel_launch(void* const* d_in, const int* in_sizes, int n_in,
                              void* d_out, int out_size, void* d_ws, size_t ws_size,
                              hipStream_t stream) {
  (void)in_sizes; (void)n_in; (void)out_size; (void)ws_size;
  const float* x          = (const float*)d_in[0];
  const int*   edge_index = (const int*)  d_in[1];
  const float* edge_attr  = (const float*)d_in[2];
  const int*   pairs      = (const int*)  d_in[3];
  const float* W1   = (const float*)d_in[4];
  const float* as1  = (const float*)d_in[5];
  const float* ad1  = (const float*)d_in[6];
  const float* We1  = (const float*)d_in[7];
  const float* ae1w = (const float*)d_in[8];
  const float* b1   = (const float*)d_in[9];
  const float* W2   = (const float*)d_in[10];
  const float* as2  = (const float*)d_in[11];
  const float* ad2  = (const float*)d_in[12];
  const float* We2  = (const float*)d_in[13];
  const float* ae2w = (const float*)d_in[14];
  const float* b2   = (const float*)d_in[15];
  const float* Wc1  = (const float*)d_in[16];
  const float* bc1  = (const float*)d_in[17];
  const float* Wc2  = (const float*)d_in[18];
  const float* bc2  = (const float*)d_in[19];
  float* logits = (float*)d_out;

  // ---- bump allocator over d_ws (256B aligned) ----
  size_t off = 0;
  auto alloc = [&](size_t bytes) -> char* {
    char* p = (char*)d_ws + off;
    off = (off + bytes + 255) & ~(size_t)255;
    return p;
  };
  int*      src   = (int*)     alloc((size_t)ETOT * 4);
  int*      dst   = (int*)     alloc((size_t)ETOT * 4);
  float*    ea    = (float*)   alloc((size_t)ETOT * FE_DIM * 4);
  float*    cnt   = (float*)   alloc((size_t)N_NODES * 4);
  float*    easum = (float*)   alloc((size_t)N_NODES * FE_DIM * 4); // reused as al/ar
  _Float16* x_h   = (_Float16*)alloc((size_t)N_NODES * FN_DIM * 2);
  _Float16* W1t_h = (_Float16*)alloc((size_t)HC1 * FN_DIM * 2);
  _Float16* W2t_h = (_Float16*)alloc((size_t)C_CH * HC1 * 2);
  _Float16* Wc1t_h= (_Float16*)alloc((size_t)C_CH * 128 * 2);
  float*    xh1   = (float*)   alloc((size_t)N_NODES * HC1 * 4);    // reused as xh2
  float*    alpha = (float*)   alloc((size_t)ETOT * H1_HEADS * 4);  // reused layer2
  unsigned* amax  = (unsigned*)alloc((size_t)N_NODES * H1_HEADS * 4);
  float*    den   = (float*)   alloc((size_t)N_NODES * H1_HEADS * 4);
  float*    out1  = (float*)   alloc((size_t)N_NODES * HC1 * 4);    // reused as out2
  _Float16* h1_h  = (_Float16*)alloc((size_t)N_NODES * HC1 * 2);
  _Float16* h2_h  = (_Float16*)alloc((size_t)N_NODES * C_CH * 2);
  float*    M1    = (float*)   alloc(FE_DIM * H1_HEADS * 4);
  float*    M2    = (float*)   alloc(FE_DIM * 4);
  // aliases
  float* al1 = easum;                   // N*H1 floats (easum done before al needed)
  float* ar1 = easum + N_NODES * H1_HEADS;
  float* xh2 = xh1;                     // xh1 dead after layer-1 aggregation
  float* out2 = out1;                   // out1 dead after h1_h conversion
  float* al2 = al1; float* ar2 = ar1;   // layer-2 node attention terms

  const int B = 256;

  // ---- graph construction ----
  hipMemsetAsync(cnt, 0, (size_t)N_NODES * 4, stream);
  hipMemsetAsync(easum, 0, (size_t)N_NODES * FE_DIM * 4, stream);
  build_edges_kernel<<<gridFor(E2_EDGES, B), B, 0, stream>>>(
      edge_index, edge_attr, src, dst, ea, cnt, easum);
  selfloop_kernel<<<gridFor(N_NODES, B), B, 0, stream>>>(src, dst, ea, cnt, easum);

  // ---- weight / input conversion for WMMA ----
  cvt_f32_to_f16_kernel<<<gridFor((long)N_NODES * FN_DIM, B), B, 0, stream>>>(
      x, x_h, (long)N_NODES * FN_DIM);
  transpose_cvt_kernel<<<gridFor(FN_DIM * HC1, B), B, 0, stream>>>(W1, W1t_h, FN_DIM, HC1);
  transpose_cvt_kernel<<<gridFor(HC1 * C_CH, B), B, 0, stream>>>(W2, W2t_h, HC1, C_CH);
  transpose_cvt_kernel<<<gridFor(128 * C_CH, B), B, 0, stream>>>(Wc1, Wc1t_h, 128, C_CH);

  // ======================= GAT layer 1 (H=4) ==========================
  {
    int tiles = (N_NODES / 16) * (HC1 / 16);          // 20000 waves
    gemm_wmma_kernel<<<gridFor(tiles, 8), 256, 0, stream>>>(
        x_h, W1t_h, xh1, N_NODES, FN_DIM, HC1);
    attn_node_kernel<<<gridFor(N_NODES * H1_HEADS, B), B, 0, stream>>>(
        xh1, as1, ad1, al1, ar1, H1_HEADS);
    edge_att_matrix_kernel<<<1, 32, 0, stream>>>(We1, ae1w, M1, H1_HEADS);
    init_amax_kernel<<<gridFor(N_NODES * H1_HEADS, B), B, 0, stream>>>(
        amax, N_NODES * H1_HEADS);
    alpha_kernel<<<gridFor((long)ETOT * H1_HEADS, B), B, 0, stream>>>(
        src, dst, ea, al1, ar1, M1, alpha, amax, H1_HEADS);
    hipMemsetAsync(den, 0, (size_t)N_NODES * H1_HEADS * 4, stream);
    exp_den_kernel<<<gridFor((long)ETOT * H1_HEADS, B), B, 0, stream>>>(
        dst, alpha, amax, den, H1_HEADS);
    hipMemsetAsync(out1, 0, (size_t)N_NODES * HC1 * 4, stream);
    aggregate_kernel<<<gridFor((long)ETOT * HC1, B), B, 0, stream>>>(
        src, dst, xh1, alpha, den, out1, H1_HEADS);
    bias_elu_cvt_kernel<<<gridFor((long)N_NODES * HC1, B), B, 0, stream>>>(
        out1, b1, h1_h, HC1, (long)N_NODES * HC1);
  }

  // ======================= GAT layer 2 (H=1) ==========================
  {
    int tiles = (N_NODES / 16) * (C_CH / 16);         // 5000 waves
    gemm_wmma_kernel<<<gridFor(tiles, 8), 256, 0, stream>>>(
        h1_h, W2t_h, xh2, N_NODES, HC1, C_CH);
    attn_node_kernel<<<gridFor(N_NODES, B), B, 0, stream>>>(
        xh2, as2, ad2, al2, ar2, 1);
    edge_att_matrix_kernel<<<1, 32, 0, stream>>>(We2, ae2w, M2, 1);
    init_amax_kernel<<<gridFor(N_NODES, B), B, 0, stream>>>(amax, N_NODES);
    alpha_kernel<<<gridFor((long)ETOT, B), B, 0, stream>>>(
        src, dst, ea, al2, ar2, M2, alpha, amax, 1);
    hipMemsetAsync(den, 0, (size_t)N_NODES * 4, stream);
    exp_den_kernel<<<gridFor((long)ETOT, B), B, 0, stream>>>(dst, alpha, amax, den, 1);
    hipMemsetAsync(out2, 0, (size_t)N_NODES * C_CH * 4, stream);
    aggregate_kernel<<<gridFor((long)ETOT * C_CH, B), B, 0, stream>>>(
        src, dst, xh2, alpha, den, out2, 1);
    bias_elu_cvt_kernel<<<gridFor((long)N_NODES * C_CH, B), B, 0, stream>>>(
        out2, b2, h2_h, C_CH, (long)N_NODES * C_CH);
  }

  // ======================= fused pair classifier ======================
  classifier_kernel<<<gridFor(E_EDGES / 16, 8), 256, 0, stream>>>(
      pairs, h2_h, Wc1t_h, bc1, Wc2, bc2, logits);
}
